// GatedPositionalSelfAttention_31086973289112
// MI455X (gfx1250) — matrix operand order repaired and checked by reading, not compiled, not run on hardware
//
#include <hip/hip_runtime.h>

typedef __bf16 bf16;
typedef __attribute__((ext_vector_type(16))) __bf16 v16bf;
typedef __attribute__((ext_vector_type(8)))  __bf16 bf16x8;
typedef __attribute__((ext_vector_type(4)))  __bf16 bf16x4;
typedef __attribute__((ext_vector_type(8)))  float  v8f;

#define BATCH 16
#define NTOK  784
#define DIM   768
#define NH    16
#define HD    48
#define HDP   64    // hd padded to multiple of 32 for K=32 WMMA steps
#define NP    800   // key dim padded to multiple of 32

__device__ __forceinline__ v16bf load_frag2(const bf16* p0, const bf16* p1) {
    union { v16bf v; bf16x8 h[2]; } u;
    u.h[0] = *(const bf16x8*)p0;
    u.h[1] = *(const bf16x8*)p1;
    return u.v;
}

__device__ __forceinline__ v8f wmma_bf16(v16bf a, v16bf b, v8f c) {
    return __builtin_amdgcn_wmma_f32_16x16x32_bf16(false, a, false, b, (short)0, c, false, false);
}

// CDNA5 async global->LDS copy (16B per lane), tracked by ASYNCcnt.
__device__ __forceinline__ void async_copy16(void* lds_dst, const bf16* gsrc) {
    unsigned lds_off = (unsigned)(unsigned long long)lds_dst;   // low 32 bits = LDS offset
    unsigned long long gaddr = (unsigned long long)gsrc;
    asm volatile("global_load_async_to_lds_b128 %0, %1, off"
                 :: "v"(lds_off), "v"(gaddr) : "memory");
}

// Fenced wait: nothing (esp. fragment ds_loads) may be scheduled across it.
__device__ __forceinline__ void wait_async0() {
    __builtin_amdgcn_sched_barrier(0);
#if __has_builtin(__builtin_amdgcn_s_wait_asynccnt)
    __builtin_amdgcn_s_wait_asynccnt(0);
#else
    asm volatile("s_wait_asynccnt 0" ::: "memory");
#endif
    __builtin_amdgcn_sched_barrier(0);
}

__device__ __forceinline__ void sync_fenced() {
    __syncthreads();
    __builtin_amdgcn_sched_barrier(0);
}

// ---------------------------------------------------------------- zero scratch
__global__ void zero_ws(uint4* p, size_t n) {
    size_t i = (size_t)blockIdx.x * blockDim.x + threadIdx.x;
    size_t stride = (size_t)gridDim.x * blockDim.x;
    uint4 z = {0u, 0u, 0u, 0u};
    for (; i < n; i += stride) p[i] = z;
}

// -------------------------------------------------- one-time f32->bf16 of x
__global__ void convert_x(const float* __restrict__ X, bf16* __restrict__ Xb, size_t n4) {
    size_t i = (size_t)blockIdx.x * blockDim.x + threadIdx.x;
    size_t stride = (size_t)gridDim.x * blockDim.x;
    for (; i < n4; i += stride) {
        float4 f = ((const float4*)X)[i];
        bf16x4 v;
        v[0] = (bf16)f.x; v[1] = (bf16)f.y; v[2] = (bf16)f.z; v[3] = (bf16)f.w;
        ((bf16x4*)Xb)[i] = v;
    }
}

// ---------------------- one-time weight f32->bf16 + transpose: Wt[n][k] = W[k][n]
__global__ __launch_bounds__(256) void convert_wt(const float* __restrict__ W,
                                                  bf16* __restrict__ Wt) {
    __shared__ float tile[32][33];
    const int t = threadIdx.x;
    const int kb0 = blockIdx.x * 32, nb0 = blockIdx.y * 32;
    #pragma unroll
    for (int i = 0; i < 4; i++) {
        int idx = t + i*256;
        int r = idx >> 5, c = idx & 31;
        tile[r][c] = W[(size_t)(kb0 + r)*DIM + nb0 + c];
    }
    __syncthreads();
    int n = t >> 3, k4 = (t & 7) * 4;
    bf16x4 v;
    v[0] = (bf16)tile[k4+0][n]; v[1] = (bf16)tile[k4+1][n];
    v[2] = (bf16)tile[k4+2][n]; v[3] = (bf16)tile[k4+3][n];
    *(bf16x4*)&Wt[(size_t)(nb0 + n)*DIM + kb0 + k4] = v;
}

// ------------------------------------------------- positional softmax (× gate)
__global__ __launch_bounds__(256) void pos_softmax(
    const float* __restrict__ rel, const float* __restrict__ Wp,
    const float* __restrict__ bp, const float* __restrict__ gates,
    float* __restrict__ posg)
{
    __shared__ float red[8];
    const int t = threadIdx.x;
    const int row = blockIdx.x;
    const int h   = blockIdx.y;
    const float w0 = Wp[h*3+0], w1 = Wp[h*3+1], w2 = Wp[h*3+2], bb = bp[h];

    float l[4];
    float mx = -3.0e38f;
    #pragma unroll
    for (int j = 0; j < 4; j++) {
        int m = t + j*256;
        float v = -3.0e38f;
        if (m < NTOK) {
            const float* r3 = &rel[((size_t)row*NTOK + m)*3];
            v = r3[0]*w0 + r3[1]*w1 + r3[2]*w2 + bb;
        }
        l[j] = v; mx = fmaxf(mx, v);
    }
    for (int o = 16; o; o >>= 1) mx = fmaxf(mx, __shfl_xor(mx, o, 32));
    if ((t & 31) == 0) red[t >> 5] = mx;
    __syncthreads();
    float bmx = red[0];
    #pragma unroll
    for (int i = 1; i < 8; i++) bmx = fmaxf(bmx, red[i]);

    float s = 0.f;
    #pragma unroll
    for (int j = 0; j < 4; j++) {
        int m = t + j*256;
        if (m < NTOK) { l[j] = __expf(l[j] - bmx); s += l[j]; }
    }
    for (int o = 16; o; o >>= 1) s += __shfl_xor(s, o, 32);
    __syncthreads();
    if ((t & 31) == 0) red[t >> 5] = s;
    __syncthreads();
    float bs = 0.f;
    #pragma unroll
    for (int i = 0; i < 8; i++) bs += red[i];

    const float gh  = 1.f / (1.f + __expf(-gates[h]));
    const float inv = gh / bs;
    float* outp = &posg[((size_t)h*NTOK + row)*NTOK];
    #pragma unroll
    for (int j = 0; j < 4; j++) {
        int m = t + j*256;
        if (m < NTOK) outp[m] = l[j] * inv;
    }
}

// ------------------------------------------------------------ QKV projections
// Block tile 128x128, wave tile 32x64 (2x4 WMMA subtiles), async double-buffered.
// vmode 0: out[b][head][n][hdp]; vmode 1: out[b][head][hdp][NP] (V^T)
__global__ __launch_bounds__(256) void qkv_gemm(
    const bf16* __restrict__ X, const bf16* __restrict__ Wt,
    bf16* __restrict__ out, float scale, int vmode)
{
    __shared__ __align__(16) bf16 At[2][128*32];
    __shared__ __align__(16) bf16 Bt[2][128*32];

    const int t = threadIdx.x;
    const int lane = t & 31, w = t >> 5;
    const int l16 = lane & 15, lh = lane >> 4;
    const int mBase = blockIdx.x * 128;
    const int nBase = blockIdx.y * 128;
    const int wm = (w >> 1) * 32;   // M offset within 128
    const int wn = (w & 1) * 64;    // N offset within 128

    auto issue = [&](int buf, int kb) {
        #pragma unroll
        for (int i = 0; i < 2; i++) {                  // A: 128x32 = 512 x 16B
            int idx = t + i*256;
            int row = idx >> 2, c8 = (idx & 3) * 8;
            async_copy16(&At[buf][row*32 + c8],
                         &X[(size_t)(mBase + row)*DIM + kb + c8]);
        }
        #pragma unroll
        for (int i = 0; i < 2; i++) {                  // B: 128x32 = 512 x 16B
            int idx = t + i*256;
            int n = idx >> 2, c8 = (idx & 3) * 8;
            async_copy16(&Bt[buf][n*32 + c8],
                         &Wt[(size_t)(nBase + n)*DIM + kb + c8]);
        }
    };

    v8f acc[2][4];
    #pragma unroll
    for (int i = 0; i < 2; i++)
        #pragma unroll
        for (int j = 0; j < 4; j++) acc[i][j] = {};

    issue(0, 0);
    int buf = 0;
    #pragma unroll
    for (int kb = 0; kb < DIM; kb += 32) {
        wait_async0();
        sync_fenced();
        if (kb + 32 < DIM) issue(buf ^ 1, kb + 32);
        v16bf bfrag[4];
        #pragma unroll
        for (int ni = 0; ni < 4; ni++) {
            const bf16* br = &Bt[buf][(wn + ni*16 + l16)*32 + lh*8];
            bfrag[ni] = load_frag2(br, br + 16);
        }
        #pragma unroll
        for (int mi = 0; mi < 2; mi++) {
            const bf16* ar = &At[buf][(wm + mi*16 + l16)*32 + lh*8];
            v16bf a = load_frag2(ar, ar + 16);
            #pragma unroll
            for (int ni = 0; ni < 4; ni++)
                acc[mi][ni] = wmma_bf16(a, bfrag[ni], acc[mi][ni]);
        }
        buf ^= 1;
    }

    #pragma unroll
    for (int mi = 0; mi < 2; mi++)
        #pragma unroll
        for (int ni = 0; ni < 4; ni++)
            #pragma unroll
            for (int r = 0; r < 8; r++) {
                int gm = mBase + wm + mi*16 + r + lh*8;
                int gn = nBase + wn + ni*16 + l16;
                int bb = gm / NTOK, nn = gm - bb*NTOK;
                int hh = gn / HD,   hd = gn - hh*HD;
                float val = acc[mi][ni][r] * scale;
                if (vmode == 0)
                    out[(((size_t)bb*NH + hh)*NTOK + nn)*HDP + hd] = (bf16)val;
                else
                    out[(((size_t)bb*NH + hh)*HDP + hd)*NP + nn] = (bf16)val;
            }
}

// -------------------------------------------------------------- attention core
__global__ __launch_bounds__(256) void attn_kernel(
    const bf16* __restrict__ Q, const bf16* __restrict__ K,
    const bf16* __restrict__ Vt, const float* __restrict__ posg,
    const float* __restrict__ gates, bf16* __restrict__ attn)
{
    __shared__ __align__(16) bf16  Qt[16*HDP];
    __shared__ __align__(16) float S[16*NP];       // aliased as bf16 sim after softmax
    __shared__ __align__(16) float Op[8][256];
    bf16* Sb = (bf16*)S;

    const int t = threadIdx.x;
    const int lane = t & 31, w = t >> 5;
    const int l16 = lane & 15, lh = lane >> 4;
    const int qtile = blockIdx.x;                  // 0..48
    const int bh = blockIdx.y;                     // 0..255
    const int b = bh >> 4, h = bh & 15;
    const int qbase = qtile * 16;

    const bf16* Qrow = &Q[((size_t)bh*NTOK + qbase)*HDP];
    const bf16* Kb   = &K[(size_t)bh*NTOK*HDP];
    const bf16* Vb   = &Vt[(size_t)bh*HDP*NP];

    if (t < 128) {   // stage Q tile 16x64 = 128 x 16B, async (waves 0-3, no divergence)
        int row = t >> 3, c8 = (t & 7) * 8;
        async_copy16(&Qt[row*HDP + c8], &Qrow[(size_t)row*HDP + c8]);
    }
    wait_async0();
    sync_fenced();

    // phase 1: S[16 x 784] = Qt @ K^T  (B fragments straight from L2-hot K)
    for (int m16 = w; m16 < 49; m16 += 8) {
        int kb = m16 * 16;
        v8f c = {};
        #pragma unroll
        for (int hs = 0; hs < 2; hs++) {
            const bf16* ar = &Qt[l16*HDP + hs*32 + lh*8];
            v16bf a = load_frag2(ar, ar + 16);
            const bf16* br = &Kb[(size_t)(kb + l16)*HDP + hs*32 + lh*8];
            v16bf bfr = load_frag2(br, br + 16);
            c = wmma_bf16(a, bfr, c);
        }
        #pragma unroll
        for (int r = 0; r < 8; r++)
            S[(r + lh*8)*NP + kb + l16] = c[r];
    }
    __syncthreads();

    // phase 2: f32 softmax (2 rows per wave, values in registers), gate blend
    {
        const int r0 = w*2, r1 = r0 + 1;
        float e0[25], e1[25];
        float m0 = -3.0e38f, m1 = -3.0e38f;
        #pragma unroll
        for (int j = 0; j < 25; j++) {
            int m = lane + j*32;
            float a  = (m < NTOK) ? S[r0*NP + m] : -3.0e38f;
            float bq = (m < NTOK) ? S[r1*NP + m] : -3.0e38f;
            e0[j] = a; e1[j] = bq;
            m0 = fmaxf(m0, a); m1 = fmaxf(m1, bq);
        }
        for (int o = 16; o; o >>= 1) {
            m0 = fmaxf(m0, __shfl_xor(m0, o, 32));
            m1 = fmaxf(m1, __shfl_xor(m1, o, 32));
        }
        float s0 = 0.f, s1 = 0.f;
        #pragma unroll
        for (int j = 0; j < 25; j++) {
            int m = lane + j*32;
            float a  = (m < NTOK) ? __expf(e0[j] - m0) : 0.f;
            float bq = (m < NTOK) ? __expf(e1[j] - m1) : 0.f;
            e0[j] = a; e1[j] = bq; s0 += a; s1 += bq;
        }
        for (int o = 16; o; o >>= 1) {
            s0 += __shfl_xor(s0, o, 32);
            s1 += __shfl_xor(s1, o, 32);
        }
        const float gh = 1.f / (1.f + __expf(-gates[h]));
        const float i0 = (1.f - gh) / s0, i1 = (1.f - gh) / s1;
        const float* p0 = &posg[((size_t)h*NTOK + qbase + r0)*NTOK];
        const float* p1 = p0 + NTOK;
        sync_fenced();     // all waves done reading S before bf16 alias overwrite
        #pragma unroll
        for (int j = 0; j < 25; j++) {
            int m = lane + j*32;
            if (m < NTOK) {
                Sb[r0*NP + m] = (bf16)(e0[j]*i0 + p0[m]);
                Sb[r1*NP + m] = (bf16)(e1[j]*i1 + p1[m]);
            } else {
                Sb[r0*NP + m] = (bf16)0.f;   // pad 784..799 so K-steps are full
                Sb[r1*NP + m] = (bf16)0.f;
            }
        }
    }
    sync_fenced();

    // phase 3: O[16 x 64] = sim @ V^T, 25 K-steps split across wave pairs
    {
        const int ct = w & 3, kh = w >> 2;
        const int cb = ct * 16;
        const int k0 = kh ? 13 : 0, k1 = kh ? 25 : 13;
        v8f o = {};
        for (int ks = k0; ks < k1; ks++) {
            int kb = ks * 32;
            const bf16* ar = &Sb[l16*NP + kb + lh*8];
            v16bf a = load_frag2(ar, ar + 16);
            const bf16* br = &Vb[(size_t)(cb + l16)*NP + kb + lh*8];
            v16bf bfr = load_frag2(br, br + 16);
            o = wmma_bf16(a, bfr, o);
        }
        #pragma unroll
        for (int r = 0; r < 8; r++)
            Op[w][(r + lh*8)*16 + l16] = o[r];
    }
    __syncthreads();

    #pragma unroll
    for (int i = 0; i < 4; i++) {
        int idx = t + i*256;
        int row = idx >> 6, col = idx & 63;
        int cti = col >> 4, cl = col & 15;
        float val = Op[cti][row*16 + cl] + Op[cti + 4][row*16 + cl];
        if (col < HD)
            attn[((size_t)b*NTOK + qbase + row)*DIM + h*HD + col] = (bf16)val;
    }
}

// ------------------------------------------------------------ output projection
__global__ __launch_bounds__(256) void out_gemm(
    const bf16* __restrict__ A, const bf16* __restrict__ Wt,
    const float* __restrict__ bias, float* __restrict__ Out)
{
    __shared__ __align__(16) bf16 At[2][128*32];
    __shared__ __align__(16) bf16 Bt[2][128*32];

    const int t = threadIdx.x;
    const int lane = t & 31, w = t >> 5;
    const int l16 = lane & 15, lh = lane >> 4;
    const int mBase = blockIdx.x * 128;
    const int nBase = blockIdx.y * 128;
    const int wm = (w >> 1) * 32;
    const int wn = (w & 1) * 64;

    auto issue = [&](int buf, int kb) {
        #pragma unroll
        for (int i = 0; i < 2; i++) {
            int idx = t + i*256;
            int row = idx >> 2, c8 = (idx & 3) * 8;
            async_copy16(&At[buf][row*32 + c8],
                         &A[(size_t)(mBase + row)*DIM + kb + c8]);
        }
        #pragma unroll
        for (int i = 0; i < 2; i++) {
            int idx = t + i*256;
            int n = idx >> 2, c8 = (idx & 3) * 8;
            async_copy16(&Bt[buf][n*32 + c8],
                         &Wt[(size_t)(nBase + n)*DIM + kb + c8]);
        }
    };

    v8f acc[2][4];
    #pragma unroll
    for (int i = 0; i < 2; i++)
        #pragma unroll
        for (int j = 0; j < 4; j++) acc[i][j] = {};

    issue(0, 0);
    int buf = 0;
    #pragma unroll
    for (int kb = 0; kb < DIM; kb += 32) {
        wait_async0();
        sync_fenced();
        if (kb + 32 < DIM) issue(buf ^ 1, kb + 32);
        v16bf bfrag[4];
        #pragma unroll
        for (int ni = 0; ni < 4; ni++) {
            const bf16* br = &Bt[buf][(wn + ni*16 + l16)*32 + lh*8];
            bfrag[ni] = load_frag2(br, br + 16);
        }
        #pragma unroll
        for (int mi = 0; mi < 2; mi++) {
            const bf16* ar = &At[buf][(wm + mi*16 + l16)*32 + lh*8];
            v16bf a = load_frag2(ar, ar + 16);
            #pragma unroll
            for (int ni = 0; ni < 4; ni++)
                acc[mi][ni] = wmma_bf16(a, bfrag[ni], acc[mi][ni]);
        }
        buf ^= 1;
    }

    #pragma unroll
    for (int mi = 0; mi < 2; mi++)
        #pragma unroll
        for (int ni = 0; ni < 4; ni++)
            #pragma unroll
            for (int r = 0; r < 8; r++) {
                int gm = mBase + wm + mi*16 + r + lh*8;
                int gn = nBase + wn + ni*16 + l16;
                Out[(size_t)gm*DIM + gn] = acc[mi][ni][r] + bias[gn];
            }
}

// -------------------------------------------------------------------- launcher
extern "C" void kernel_launch(void* const* d_in, const int* in_sizes, int n_in,
                              void* d_out, int out_size, void* d_ws, size_t ws_size,
                              hipStream_t stream) {
    const float* x     = (const float*)d_in[0];
    const float* Wq    = (const float*)d_in[1];
    const float* Wk    = (const float*)d_in[2];
    const float* Wv    = (const float*)d_in[3];
    const float* Wp    = (const float*)d_in[4];
    const float* bp    = (const float*)d_in[5];
    const float* Wout  = (const float*)d_in[6];
    const float* bout  = (const float*)d_in[7];
    const float* gates = (const float*)d_in[8];
    const float* rel   = (const float*)d_in[9];
    float* out = (float*)d_out;
    char* ws = (char*)d_ws;

    const size_t SZ_POS = (size_t)NH * NTOK * NTOK * 4;          // 39.3 MB
    const size_t SZ_QK  = (size_t)BATCH * NH * NTOK * HDP * 2;   // 25.7 MB
    const size_t SZ_V   = (size_t)BATCH * NH * HDP * NP * 2;     // 26.2 MB
    const size_t SZ_A   = (size_t)BATCH * NTOK * DIM * 2;        // 19.3 MB
    const size_t SZ_W   = (size_t)DIM * DIM * 2;                 //  1.2 MB
    const size_t OFF_POS = 0;
    const size_t OFF_Q   = OFF_POS + SZ_POS;
    const size_t OFF_K   = OFF_Q + SZ_QK;
    const size_t OFF_V   = OFF_K + SZ_QK;
    const size_t OFF_A   = OFF_V + SZ_V;
    const size_t OFF_XB  = OFF_A + SZ_A;
    const size_t OFF_WTQ = OFF_XB + SZ_A;
    const size_t OFF_WTK = OFF_WTQ + SZ_W;
    const size_t OFF_WTV = OFF_WTK + SZ_W;
    const size_t OFF_WTO = OFF_WTV + SZ_W;

    convert_x<<<dim3(4096), dim3(256), 0, stream>>>(
        x, (bf16*)(ws + OFF_XB), (size_t)BATCH*NTOK*DIM/4);
    convert_wt<<<dim3(24, 24), dim3(256), 0, stream>>>(Wq, (bf16*)(ws + OFF_WTQ));
    convert_wt<<<dim3(24, 24), dim3(256), 0, stream>>>(Wk, (bf16*)(ws + OFF_WTK));
    convert_wt<<<dim3(24, 24), dim3(256), 0, stream>>>(Wv, (bf16*)(ws + OFF_WTV));
    convert_wt<<<dim3(24, 24), dim3(256), 0, stream>>>(Wout, (bf16*)(ws + OFF_WTO));

    zero_ws<<<dim3(2048), dim3(256), 0, stream>>>(
        (uint4*)(ws + OFF_Q), (2*SZ_QK + SZ_V) / 16);

    pos_softmax<<<dim3(NTOK, NH), dim3(256), 0, stream>>>(
        rel, Wp, bp, gates, (float*)(ws + OFF_POS));

    const float scale = 0.14433756729740643f;  // 48^-0.5
    qkv_gemm<<<dim3(98, 6), dim3(256), 0, stream>>>(
        (const bf16*)(ws + OFF_XB), (const bf16*)(ws + OFF_WTQ), (bf16*)(ws + OFF_Q), scale, 0);
    qkv_gemm<<<dim3(98, 6), dim3(256), 0, stream>>>(
        (const bf16*)(ws + OFF_XB), (const bf16*)(ws + OFF_WTK), (bf16*)(ws + OFF_K), 1.0f, 0);
    qkv_gemm<<<dim3(98, 6), dim3(256), 0, stream>>>(
        (const bf16*)(ws + OFF_XB), (const bf16*)(ws + OFF_WTV), (bf16*)(ws + OFF_V), 1.0f, 1);

    attn_kernel<<<dim3(49, BATCH * NH), dim3(256), 0, stream>>>(
        (const bf16*)(ws + OFF_Q), (const bf16*)(ws + OFF_K),
        (const bf16*)(ws + OFF_V), (const float*)(ws + OFF_POS),
        gates, (bf16*)(ws + OFF_A));

    out_gemm<<<dim3(98, 6), dim3(256), 0, stream>>>(
        (const bf16*)(ws + OFF_A), (const bf16*)(ws + OFF_WTO), bout, out);
}
